// StochasticKMeans_73400991089049
// MI455X (gfx1250) — matrix operand order repaired and checked by reading, not compiled, not run on hardware
//
#include <hip/hip_runtime.h>
#include <hip/hip_bf16.h>

typedef __attribute__((ext_vector_type(16))) __bf16 v16bf;
typedef __attribute__((ext_vector_type(8)))  float  v8f;

#define N_PTS 131072
#define FEAT  64
#define K_CL  2048

#define CHUNK      4                    // cluster tiles per LDS chunk
#define CHUNK_CL   (CHUNK * 16)         // 64 clusters per chunk
#define CHUNK_EL   (CHUNK_CL * FEAT)    // 4096 bf16 = 8 KB
#define CHUNK_V4   (CHUNK_EL / 8)       // 512 uint4 (16B units)
#define V4_PER_THR (CHUNK_V4 / 256)     // 2 per thread
#define NCHUNKS    (K_CL / CHUNK_CL)    // 32

// f32 -> bf16 round-to-nearest-even (bit path)
static __device__ __forceinline__ __bf16 f2bf(float f) {
  unsigned u = __builtin_bit_cast(unsigned, f);
  unsigned r = u + 0x7FFFu + ((u >> 16) & 1u);
  unsigned short h = (unsigned short)(r >> 16);
  return __builtin_bit_cast(__bf16, h);
}

// One block (64 threads) per cluster: cb[k][f] = bf16(-2*c[k][f]); cnorm[k] = ||c||^2.
__global__ void prep_kernel(const float* __restrict__ c, __bf16* __restrict__ cb,
                            float* __restrict__ cnorm) {
  const int k = blockIdx.x;
  const int f = threadIdx.x;
  float v = c[k * FEAT + f];
  cb[k * FEAT + f] = f2bf(-2.0f * v);
  __shared__ float s[FEAT];
  s[f] = v * v;
  __syncthreads();
  for (int step = FEAT / 2; step > 0; step >>= 1) {
    if (f < step) s[f] += s[f + step];
    __syncthreads();
  }
  if (f == 0) cnorm[k] = s[0];
}

// 8 waves/block; each wave owns 32 points (two M-tiles sharing B fragments).
// B staged through LDS in double-buffered 8KB chunks (one barrier per chunk);
// per-tile value: ||c||^2 - 2 x.c  (||x||^2 constant per row -> dropped).
// Argmin: one cmp per element feeding both value and index selects
// (3 VALU/elem, no NaN-canonicalization overhead, no key packing).
__global__ void __launch_bounds__(256) assign_kernel(const float* __restrict__ x,
                                                     const __bf16* __restrict__ cb,
                                                     const float* __restrict__ cnorm,
                                                     int* __restrict__ assign_out) {
  __shared__ __align__(16) __bf16 sB[2][CHUNK_EL];  // 2 x 8 KB B staging
  __shared__ float sCn[K_CL];                       // 8 KB ||c||^2

  const int tid  = threadIdx.x;
  const int lane = tid & 31;
  const int wave = tid >> 5;
  const int col  = lane & 15;   // N column (and M row) within tile
  const int half = lane >> 4;
  const int row0 = blockIdx.x * 256 + wave * 32;

  const uint4* gB = (const uint4*)cb;  // B in 16-byte units

  // stage all cnorms + chunk 0 of B; one barrier covers both
  for (int i = tid; i < K_CL; i += 256) sCn[i] = cnorm[i];
#pragma unroll
  for (int j = 0; j < V4_PER_THR; ++j)
    ((uint4*)sB[0])[j * 256 + tid] = gB[j * 256 + tid];

  // A fragments: lanes 0-15 hold K=0..15 contiguous, lanes 16-31 K=16..31
  // (matches B order).
  const int kb = half * 16;
  const float* xr0 = x + (size_t)(row0 + col) * FEAT;
  const float* xr1 = x + (size_t)(row0 + 16 + col) * FEAT;
  v16bf a0r0, a1r0, a0r1, a1r1;
#pragma unroll
  for (int i = 0; i < 8; ++i) {
    a0r0[i]     = f2bf(xr0[kb + i]);
    a0r0[8 + i] = f2bf(xr0[8 + kb + i]);
    a1r0[i]     = f2bf(xr0[32 + kb + i]);
    a1r0[8 + i] = f2bf(xr0[40 + kb + i]);
    a0r1[i]     = f2bf(xr1[kb + i]);
    a0r1[8 + i] = f2bf(xr1[8 + kb + i]);
    a1r1[i]     = f2bf(xr1[32 + kb + i]);
    a1r1[8 + i] = f2bf(xr1[40 + kb + i]);
  }

  float minv0[8], minv1[8];
  int   mini0[8], mini1[8];
#pragma unroll
  for (int r = 0; r < 8; ++r) {
    minv0[r] = 3.4e38f; mini0[r] = 0;
    minv1[r] = 3.4e38f; mini1[r] = 0;
  }

  __syncthreads();  // sCn + chunk 0 visible

  // B layout: lanes 0-15 hold K=0..15 contiguous, lanes 16-31 K=16..31.
  const int kb2 = half * 16;

  for (int ch = 0; ch < NCHUNKS; ++ch) {
    const int p = ch & 1;

    // kick off next chunk's global loads (waits land at the ds_stores below)
    uint4 nx[V4_PER_THR];
    const int nch = (ch + 1) & (NCHUNKS - 1);  // wraps harmlessly on last chunk
    const uint4* g = gB + (size_t)nch * CHUNK_V4;
#pragma unroll
    for (int j = 0; j < V4_PER_THR; ++j) nx[j] = g[j * 256 + tid];

    const __bf16* lbase = sB[p];
#pragma unroll
    for (int t = 0; t < CHUNK; ++t) {
      const int nt = ch * CHUNK + t;
      const __bf16* crow = lbase + (t * 16 + col) * FEAT + kb2;
      v16bf b0 = *(const v16bf*)(crow);        // ds_load_b128 x2
      v16bf b1 = *(const v16bf*)(crow + 32);
      const float cn = sCn[nt * 16 + col];     // ds_load_b32

      v8f acc0, acc1;
#pragma unroll
      for (int r = 0; r < 8; ++r) { acc0[r] = cn; acc1[r] = cn; }  // seed C=||c||^2

      // interleave accumulators: no WMMA->WMMA RAW on D
      acc0 = __builtin_amdgcn_wmma_f32_16x16x32_bf16(false, a0r0, false, b0,
                                                     (short)0, acc0, false, false);
      acc1 = __builtin_amdgcn_wmma_f32_16x16x32_bf16(false, a0r1, false, b0,
                                                     (short)0, acc1, false, false);
      acc0 = __builtin_amdgcn_wmma_f32_16x16x32_bf16(false, a1r0, false, b1,
                                                     (short)0, acc0, false, false);
      acc1 = __builtin_amdgcn_wmma_f32_16x16x32_bf16(false, a1r1, false, b1,
                                                     (short)0, acc1, false, false);

      const int myc = nt * 16 + col;
#pragma unroll
      for (int r = 0; r < 8; ++r) {
        const bool lt0 = acc0[r] < minv0[r];
        const bool lt1 = acc1[r] < minv1[r];
        mini0[r] = lt0 ? myc : mini0[r];          // v_cndmask (shares cmp)
        minv0[r] = lt0 ? acc0[r] : minv0[r];      // v_cndmask
        mini1[r] = lt1 ? myc : mini1[r];
        minv1[r] = lt1 ? acc1[r] : minv1[r];
      }
    }

    // commit next chunk to the other buffer; barrier ends this chunk
#pragma unroll
    for (int j = 0; j < V4_PER_THR; ++j)
      ((uint4*)sB[p ^ 1])[j * 256 + tid] = nx[j];
    __syncthreads();
  }

  // Reduce argmin across the 16 columns in each 16-lane half (xor shuffles).
#pragma unroll
  for (int m = 1; m < 16; m <<= 1) {
#pragma unroll
    for (int r = 0; r < 8; ++r) {
      float ov0 = __shfl_xor(minv0[r], m, 32);
      int   oi0 = __shfl_xor(mini0[r], m, 32);
      if (ov0 < minv0[r] || (ov0 == minv0[r] && oi0 < mini0[r])) {
        minv0[r] = ov0; mini0[r] = oi0;
      }
      float ov1 = __shfl_xor(minv1[r], m, 32);
      int   oi1 = __shfl_xor(mini1[r], m, 32);
      if (ov1 < minv1[r] || (ov1 == minv1[r] && oi1 < mini1[r])) {
        minv1[r] = ov1; mini1[r] = oi1;
      }
    }
  }
  // D layout: VGPR r -> row r (lanes 0-15) / row 8+r (lanes 16-31)
  if (col == 0) {
    const int base0 = row0 + half * 8;
    const int base1 = row0 + 16 + half * 8;
#pragma unroll
    for (int r = 0; r < 8; ++r) {
      assign_out[base0 + r] = mini0[r];
      assign_out[base1 + r] = mini1[r];
    }
  }
}

// Cluster-partitioned scatter-mean: 128 blocks x 16 clusters each. Every block
// streams all assignments, accumulates matching points into LDS (ds_add_f32),
// then WRITES its slice of sums/counts. Zero global atomics.
#define SCAT_BLOCKS 128
#define CPB (K_CL / SCAT_BLOCKS)  // 16 clusters per block

__global__ void __launch_bounds__(256) scatter_kernel(const float* __restrict__ x,
                                                      const int* __restrict__ assign,
                                                      float* __restrict__ sums,
                                                      float* __restrict__ counts) {
  __shared__ float ssum[CPB * FEAT];  // 4 KB
  __shared__ float scnt[CPB];
  const int tid = threadIdx.x;
  const int c0  = blockIdx.x * CPB;
  for (int i = tid; i < CPB * FEAT; i += 256) ssum[i] = 0.0f;
  if (tid < CPB) scnt[tid] = 0.0f;
  __syncthreads();

  for (int p = tid; p < N_PTS; p += 256) {
    const int a = assign[p] - c0;
    if ((unsigned)a < (unsigned)CPB) {
      const float4* xr = (const float4*)(x + (size_t)p * FEAT);
      atomicAdd(&scnt[a], 1.0f);
#pragma unroll
      for (int q = 0; q < FEAT / 4; ++q) {
        float4 v = xr[q];
        atomicAdd(&ssum[a * FEAT + 4 * q + 0], v.x);
        atomicAdd(&ssum[a * FEAT + 4 * q + 1], v.y);
        atomicAdd(&ssum[a * FEAT + 4 * q + 2], v.z);
        atomicAdd(&ssum[a * FEAT + 4 * q + 3], v.w);
      }
    }
  }
  __syncthreads();

  for (int i = tid; i < CPB * FEAT; i += 256) sums[c0 * FEAT + i] = ssum[i];
  if (tid < CPB) counts[c0 + tid] = scnt[tid];
}

// EMA update: out = 0.99*c + 0.01*(count>0 ? sum/count : c)
__global__ void update_kernel(const float* __restrict__ c, const float* __restrict__ sums,
                              const float* __restrict__ counts, float* __restrict__ out) {
  const int t = blockIdx.x * blockDim.x + threadIdx.x;
  const int k = t >> 6;
  const float cnt  = counts[k];
  const float cv   = c[t];
  const float mean = sums[t] / fmaxf(cnt, 1.0f);
  const float nc   = cnt > 0.0f ? mean : cv;
  out[t] = 0.99f * cv + 0.01f * nc;
}

extern "C" void kernel_launch(void* const* d_in, const int* in_sizes, int n_in,
                              void* d_out, int out_size, void* d_ws, size_t ws_size,
                              hipStream_t stream) {
  const float* x = (const float*)d_in[0];  // [N, FEAT] f32
  const float* c = (const float*)d_in[1];  // [K, FEAT] f32

  char* ws = (char*)d_ws;
  __bf16* cb   = (__bf16*)ws;                                    // 256 KB bf16(-2c)
  float* cnorm = (float*)(ws + (size_t)K_CL * FEAT * 2);         // 8 KB
  float* sums  = (float*)(ws + (size_t)K_CL * FEAT * 2 + 8192);  // 512 KB
  float* cnts  = sums + (size_t)K_CL * FEAT;                     // 8 KB

  int*   assign_out = (int*)d_out;            // N ints
  float* updated    = (float*)d_out + N_PTS;  // K*FEAT f32

  prep_kernel<<<K_CL, FEAT, 0, stream>>>(c, cb, cnorm);
  assign_kernel<<<N_PTS / 256, 256, 0, stream>>>(x, cb, cnorm, assign_out);
  scatter_kernel<<<SCAT_BLOCKS, 256, 0, stream>>>(x, assign_out, sums, cnts);
  update_kernel<<<(K_CL * FEAT) / 256, 256, 0, stream>>>(c, sums, cnts, updated);
}
